// ReprojLikelihood_29944511988266
// MI455X (gfx1250) — matrix operand order repaired and checked by reading, not compiled, not run on hardware
//
#include <hip/hip_runtime.h>
#include <hip/hip_bf16.h>

// Problem constants (reference: B, M, N, D = 4, 4096, 4096, 256)
#define BB 4
#define MM 4096
#define NN 4096
#define DD 256

typedef __bf16 v16bf __attribute__((ext_vector_type(16)));
typedef float  v8f   __attribute__((ext_vector_type(8)));

union FragBF {
    v16bf v;
    uint4 q[2];
};

#ifndef __has_builtin
#define __has_builtin(x) 0
#endif
#if __has_builtin(__builtin_amdgcn_tensor_load_to_lds) && \
    __has_builtin(__builtin_amdgcn_s_wait_tensorcnt)
#define USE_TDM 1
#else
#define USE_TDM 0
#endif

__device__ __forceinline__ unsigned short f2bf(float x) {
    unsigned int u = __float_as_uint(x);
    u += 0x7FFFu + ((u >> 16) & 1u);   // round-to-nearest-even
    return (unsigned short)(u >> 16);
}

__device__ __forceinline__ v8f wmma_bf16(const v16bf& a, const v16bf& b, const v8f& c) {
    return __builtin_amdgcn_wmma_f32_16x16x32_bf16(false, a, false, b, (short)0, c,
                                                   false, false);
}

#if USE_TDM
typedef unsigned int v4u __attribute__((ext_vector_type(4)));
typedef int          v8i __attribute__((ext_vector_type(8)));
typedef int          v4i __attribute__((ext_vector_type(4)));

// 2-D bf16 tile DMA: global (row stride = stride_el elements) -> LDS packed.
// D# layout per CDNA5 ISA: group0 = {count=1,
// lds_addr, global_addr, type=2}; group1 = {data_size=2B, tensor_dim0/1,
// tile_dim0/1, tensor_dim0_stride}. Groups 2/3 unused (2-D tensor).
__device__ __forceinline__ void tdm_load_2d(const void* lds_ptr,
                                            const void* gptr,
                                            unsigned tile0_el,    // tile row length (elements)
                                            unsigned tile1_rows,  // tile rows
                                            unsigned stride_el)   // tensor row stride (elements)
{
    unsigned long long ga = (unsigned long long)(size_t)gptr;
    unsigned lds_addr = (unsigned)(size_t)lds_ptr;   // flat LDS aperture: addr[31:0]
    v4u g0;
    g0[0] = 1u;                                                  // count=1, user mode
    g0[1] = lds_addr;
    g0[2] = (unsigned)ga;                                        // global_addr[31:0]
    g0[3] = (unsigned)((ga >> 32) & 0x01FFFFFFu) | (2u << 30);   // addr[56:32] | type=2
    v8i g1;
    g1[0] = 0x10000;                                             // data_size=1 -> 2 bytes
    g1[1] = (int)((stride_el & 0xFFFFu) << 16);                  // tensor_dim0[15:0]
    g1[2] = (int)(((stride_el >> 16) & 0xFFFFu) |
                  ((tile1_rows & 0xFFFFu) << 16));               // dim0 hi | tensor_dim1 lo
    g1[3] = (int)(((tile1_rows >> 16) & 0xFFFFu) |
                  ((tile0_el & 0xFFFFu) << 16));                 // dim1 hi | tile_dim0
    g1[4] = (int)tile1_rows;                                     // tile_dim1 (tile_dim2=0)
    g1[5] = (int)stride_el;                                      // tensor_dim0_stride[31:0]
    g1[6] = 0;
    g1[7] = 0;
    v4i z4 = {0, 0, 0, 0};
#if __clang_major__ >= 23
    v8i z8 = {0, 0, 0, 0, 0, 0, 0, 0};
    __builtin_amdgcn_tensor_load_to_lds(g0, g1, z4, z4, z8, 0);
#else
    __builtin_amdgcn_tensor_load_to_lds(g0, g1, z4, z4, 0);
#endif
}
#endif  // USE_TDM

// ---------------------------------------------------------------------------
// Kernel 1: fused final_proj for desc0 and desc1.
//   md[r][e] = bf16( (desc[r,:] . Wf[e,:] + bf[e]) * 0.25 )   (scale = 256^0.25)
// Grid: (E/64 = 4, (B*(M+N))/64 = 512), 256 threads (8 waves).
// ---------------------------------------------------------------------------
__global__ __launch_bounds__(256) void proj_kernel(
    const float* __restrict__ desc0, const float* __restrict__ desc1,
    const float* __restrict__ Wf, const float* __restrict__ bfv,
    unsigned short* __restrict__ md0, unsigned short* __restrict__ md1)
{
    __shared__ __align__(16) unsigned short AL[64 * 64];  // desc chunk, bf16
    __shared__ __align__(16) unsigned short BL[64 * 64];  // Wf chunk,  bf16

    const int tid = threadIdx.x;
    const int eb  = blockIdx.x * 64;
    const int rb  = blockIdx.y;

    const float* src;
    unsigned short* dst;
    int rowOff;
    if (rb < (BB * MM) / 64) { src = desc0; dst = md0; rowOff = rb * 64; }
    else                     { src = desc1; dst = md1; rowOff = (rb - (BB * MM) / 64) * 64; }

    const int w   = tid >> 5;
    const int ln  = tid & 31;
    const int hf  = ln >> 4;
    const int l15 = ln & 15;
    const int mi  = w & 3;
    const int ni0 = w >> 2;

    v8f c0 = {0,0,0,0,0,0,0,0};
    v8f c1 = {0,0,0,0,0,0,0,0};

    for (int kc = 0; kc < 4; ++kc) {
        __syncthreads();
        #pragma unroll
        for (int i = 0; i < 4; ++i) {
            int lin = tid + 256 * i;
            int r   = lin >> 4;
            int c4  = lin & 15;
            float4 a = *(const float4*)(src + (size_t)(rowOff + r) * DD + kc * 64 + c4 * 4);
            uint2 pa;
            pa.x = (unsigned)f2bf(a.x) | ((unsigned)f2bf(a.y) << 16);
            pa.y = (unsigned)f2bf(a.z) | ((unsigned)f2bf(a.w) << 16);
            *(uint2*)(AL + r * 64 + c4 * 4) = pa;
            float4 b = *(const float4*)(Wf + (size_t)(eb + r) * DD + kc * 64 + c4 * 4);
            uint2 pb;
            pb.x = (unsigned)f2bf(b.x) | ((unsigned)f2bf(b.y) << 16);
            pb.y = (unsigned)f2bf(b.z) | ((unsigned)f2bf(b.w) << 16);
            *(uint2*)(BL + r * 64 + c4 * 4) = pb;
        }
        __syncthreads();

        const unsigned short* aRow  = AL + (mi  * 16 + l15) * 64;
        const unsigned short* bRow0 = BL + (ni0 * 16 + l15) * 64;
        const unsigned short* bRow1 = bRow0 + 32 * 64;
        FragBF fa0, fa1, fb00, fb01, fb10, fb11;
        fa0.q[0]  = *(const uint4*)(aRow  + 16 * hf);
        fa0.q[1]  = *(const uint4*)(aRow  + 16 * hf + 8);
        fa1.q[0]  = *(const uint4*)(aRow  + 32 + 16 * hf);
        fa1.q[1]  = *(const uint4*)(aRow  + 32 + 16 * hf + 8);
        fb00.q[0] = *(const uint4*)(bRow0 + 16 * hf);
        fb00.q[1] = *(const uint4*)(bRow0 + 16 * hf + 8);
        fb01.q[0] = *(const uint4*)(bRow0 + 32 + 16 * hf);
        fb01.q[1] = *(const uint4*)(bRow0 + 32 + 16 * hf + 8);
        fb10.q[0] = *(const uint4*)(bRow1 + 16 * hf);
        fb10.q[1] = *(const uint4*)(bRow1 + 16 * hf + 8);
        fb11.q[0] = *(const uint4*)(bRow1 + 32 + 16 * hf);
        fb11.q[1] = *(const uint4*)(bRow1 + 32 + 16 * hf + 8);
        c0 = wmma_bf16(fa0.v, fb00.v, c0);
        c1 = wmma_bf16(fa0.v, fb10.v, c1);
        c0 = wmma_bf16(fa1.v, fb01.v, c0);
        c1 = wmma_bf16(fa1.v, fb11.v, c1);
    }

    const int e0 = eb + ni0 * 16 + l15;
    const int e1 = e0 + 32;
    const float b0 = bfv[e0];
    const float b1 = bfv[e1];
    #pragma unroll
    for (int v = 0; v < 8; ++v) {
        int row = rowOff + mi * 16 + 8 * hf + v;
        dst[(size_t)row * DD + e0] = f2bf((c0[v] + b0) * 0.25f);
        dst[(size_t)row * DD + e1] = f2bf((c1[v] + b1) * 0.25f);
    }
}

// ---------------------------------------------------------------------------
// Shared GEMM chunk: c0/c1 += Q[64x64 chunk] x K_chunk^T for this wave.
// ---------------------------------------------------------------------------
__device__ __forceinline__ void gemm_chunk(const unsigned short* aRow, int ka,
                                           const unsigned short* kl,
                                           int ni0, int l15, int hf,
                                           v8f& c0, v8f& c1)
{
    const unsigned short* bRow0 = kl + (ni0 * 16 + l15) * 64;
    const unsigned short* bRow1 = bRow0 + 32 * 64;
    FragBF fa0, fa1, fb00, fb01, fb10, fb11;
    fa0.q[0]  = *(const uint4*)(aRow  + ka + 16 * hf);
    fa0.q[1]  = *(const uint4*)(aRow  + ka + 16 * hf + 8);
    fa1.q[0]  = *(const uint4*)(aRow  + ka + 32 + 16 * hf);
    fa1.q[1]  = *(const uint4*)(aRow  + ka + 32 + 16 * hf + 8);
    fb00.q[0] = *(const uint4*)(bRow0 + 16 * hf);
    fb00.q[1] = *(const uint4*)(bRow0 + 16 * hf + 8);
    fb01.q[0] = *(const uint4*)(bRow0 + 32 + 16 * hf);
    fb01.q[1] = *(const uint4*)(bRow0 + 32 + 16 * hf + 8);
    fb10.q[0] = *(const uint4*)(bRow1 + 16 * hf);
    fb10.q[1] = *(const uint4*)(bRow1 + 16 * hf + 8);
    fb11.q[0] = *(const uint4*)(bRow1 + 32 + 16 * hf);
    fb11.q[1] = *(const uint4*)(bRow1 + 32 + 16 * hf + 8);
    c0 = wmma_bf16(fa0.v, fb00.v, c0);
    c1 = wmma_bf16(fa0.v, fb10.v, c1);
    c0 = wmma_bf16(fa1.v, fb01.v, c0);
    c1 = wmma_bf16(fa1.v, fb11.v, c1);
}

// ---------------------------------------------------------------------------
// Kernel 2: row-softmax statistics of Q . K^T (online flash-style max/sum).
// Launched twice: (md0, md1) -> row stats; (md1, md0) -> column stats.
// Q block is LDS-persistent; K streamed in double-buffered 64x64 chunks,
// staged by the Tensor Data Mover (TENSORcnt pipelined, lookahead 1).
// ---------------------------------------------------------------------------
__global__ __launch_bounds__(256) void stats_kernel(
    const unsigned short* __restrict__ Qm, const unsigned short* __restrict__ Km,
    float* __restrict__ omax, float* __restrict__ oinv)
{
    __shared__ __align__(16) unsigned short QL[64 * 256];     // 32 KB
    __shared__ __align__(16) unsigned short KL[2][64 * 64];   // 2 x 8 KB
    __shared__ float2 sred[128];

    const int tid   = threadIdx.x;
    const int batch = blockIdx.x >> 6;
    const int mb    = (blockIdx.x & 63) * 64;
    const size_t qbase = (size_t)batch * MM + mb;
    const size_t kbase = (size_t)batch * NN;

    const int w   = tid >> 5;
    const int ln  = tid & 31;
    const int hf  = ln >> 4;
    const int l15 = ln & 15;
    const int mi  = w & 3;
    const int ni0 = w >> 2;

#if USE_TDM
    if (w == 0) {   // one DMA for the whole 64x256 Q block
        tdm_load_2d(QL, Qm + qbase * DD, 256, 64, 256);
    }
#else
    #pragma unroll
    for (int i = 0; i < 8; ++i) {
        int lin = tid + 256 * i;
        int r = lin >> 5, q = lin & 31;
        ((uint4*)QL)[lin] = ((const uint4*)(Qm + (qbase + r) * DD))[q];
    }
#endif

    const unsigned short* aRow = QL + (mi * 16 + l15) * DD;

    float rmx[8], rsm[8];
    #pragma unroll
    for (int v = 0; v < 8; ++v) { rmx[v] = -3.0e38f; rsm[v] = 0.0f; }

    for (int nb = 0; nb < NN; nb += 64) {
        v8f c0 = {0,0,0,0,0,0,0,0};
        v8f c1 = {0,0,0,0,0,0,0,0};
        const unsigned short* kg = Km + (kbase + nb) * DD;
#if USE_TDM
        if (w == 0) tdm_load_2d(KL[0], kg, 64, 64, 256);
#endif
        #pragma unroll
        for (int kc = 0; kc < 4; ++kc) {
#if USE_TDM
            if (w == 0) {
                if (kc < 3) {
                    tdm_load_2d(KL[(kc + 1) & 1], kg + (kc + 1) * 64, 64, 64, 256);
                    __builtin_amdgcn_s_wait_tensorcnt(1);
                } else {
                    __builtin_amdgcn_s_wait_tensorcnt(0);
                }
            }
#else
            #pragma unroll
            for (int i = 0; i < 2; ++i) {
                int lin = tid + 256 * i;
                int r = lin >> 3, q = lin & 7;
                ((uint4*)KL[kc & 1])[lin] = ((const uint4*)(kg + r * DD + kc * 64))[q];
            }
#endif
            __syncthreads();
            gemm_chunk(aRow, kc * 64, KL[kc & 1], ni0, l15, hf, c0, c1);
            __syncthreads();
        }
        // Online (max, sumexp) over this wave's 32 columns of the slab.
        #pragma unroll
        for (int v = 0; v < 8; ++v) {
            float t = fmaxf(c0[v], c1[v]);
            t = fmaxf(t, __shfl_xor(t, 1, 32));
            t = fmaxf(t, __shfl_xor(t, 2, 32));
            t = fmaxf(t, __shfl_xor(t, 4, 32));
            t = fmaxf(t, __shfl_xor(t, 8, 32));
            float nm = fmaxf(rmx[v], t);
            float s = __expf(c0[v] - nm) + __expf(c1[v] - nm);
            s += __shfl_xor(s, 1, 32);
            s += __shfl_xor(s, 2, 32);
            s += __shfl_xor(s, 4, 32);
            s += __shfl_xor(s, 8, 32);
            rsm[v] = rsm[v] * __expf(rmx[v] - nm) + s;
            rmx[v] = nm;
        }
    }

    if (l15 == 0) {
        #pragma unroll
        for (int v = 0; v < 8; ++v)
            sred[w * 16 + hf * 8 + v] = make_float2(rmx[v], rsm[v]);
    }
    __syncthreads();
    if (tid < 64) {
        int r = tid, miB = r >> 4, lr = r & 15;
        float2 a  = sred[miB * 16 + lr];
        float2 b2 = sred[(miB + 4) * 16 + lr];
        float Mx = fmaxf(a.x, b2.x);
        float S  = a.y * __expf(a.x - Mx) + b2.y * __expf(b2.x - Mx);
        omax[qbase + r] = Mx;
        oinv[qbase + r] = 1.0f / S;
    }
}

// ---------------------------------------------------------------------------
// Kernel 3: recompute each 64x64 sim tile once, write both softmax outputs
// with non-temporal stores (outputs are write-once; keep L2 for md0/md1).
// ---------------------------------------------------------------------------
__global__ __launch_bounds__(256) void pwrite_kernel(
    const unsigned short* __restrict__ md0, const unsigned short* __restrict__ md1,
    const float* __restrict__ rmax, const float* __restrict__ rinv,
    const float* __restrict__ cmax, const float* __restrict__ cinv,
    float* __restrict__ p01, float* __restrict__ p10)
{
    __shared__ __align__(16) unsigned short QL[64 * 256];
    __shared__ __align__(16) unsigned short KL[2][64 * 64];
    __shared__ float sRm[64], sRi[64], sCm[64], sCi[64];

    const int tid = threadIdx.x;
    const int b   = blockIdx.z;
    const int mb  = blockIdx.y * 64;
    const int nb  = blockIdx.x * 64;
    const size_t qbase = (size_t)b * MM + mb;
    const size_t kbase = (size_t)b * NN + nb;

    const int w   = tid >> 5;
    const int ln  = tid & 31;
    const int hf  = ln >> 4;
    const int l15 = ln & 15;
    const int mi  = w & 3;
    const int ni0 = w >> 2;

    if (tid < 64) {
        sRm[tid] = rmax[qbase + tid];
        sRi[tid] = rinv[qbase + tid];
        sCm[tid] = cmax[kbase + tid];
        sCi[tid] = cinv[kbase + tid];
    }

    const unsigned short* kg = md1 + kbase * DD;
#if USE_TDM
    if (w == 0) {
        tdm_load_2d(QL, md0 + qbase * DD, 256, 64, 256);
        tdm_load_2d(KL[0], kg, 64, 64, 256);
    }
#else
    #pragma unroll
    for (int i = 0; i < 8; ++i) {
        int lin = tid + 256 * i;
        int r = lin >> 5, q = lin & 31;
        ((uint4*)QL)[lin] = ((const uint4*)(md0 + (qbase + r) * DD))[q];
    }
#endif

    const unsigned short* aRow = QL + (mi * 16 + l15) * DD;

    v8f c0 = {0,0,0,0,0,0,0,0};
    v8f c1 = {0,0,0,0,0,0,0,0};
    #pragma unroll
    for (int kc = 0; kc < 4; ++kc) {
#if USE_TDM
        if (w == 0) {
            if (kc < 3) {
                tdm_load_2d(KL[(kc + 1) & 1], kg + (kc + 1) * 64, 64, 64, 256);
                __builtin_amdgcn_s_wait_tensorcnt(1);
            } else {
                __builtin_amdgcn_s_wait_tensorcnt(0);
            }
        }
#else
        #pragma unroll
        for (int i = 0; i < 2; ++i) {
            int lin = tid + 256 * i;
            int r = lin >> 3, q = lin & 7;
            ((uint4*)KL[kc & 1])[lin] = ((const uint4*)(kg + r * DD + kc * 64))[q];
        }
#endif
        __syncthreads();
        gemm_chunk(aRow, kc * 64, KL[kc & 1], ni0, l15, hf, c0, c1);
        __syncthreads();
    }

    const int nl0 = ni0 * 16 + l15;
    const int nl1 = nl0 + 32;
    const float cm0 = sCm[nl0], ci0 = sCi[nl0];
    const float cm1 = sCm[nl1], ci1 = sCi[nl1];
    #pragma unroll
    for (int v = 0; v < 8; ++v) {
        int ml = mi * 16 + 8 * hf + v;
        float rm = sRm[ml], ri = sRi[ml];
        size_t o = ((size_t)b * MM + (mb + ml)) * (size_t)NN + nb;
        float e0 = c0[v], e1 = c1[v];
        __builtin_nontemporal_store(__expf(e0 - rm) * ri, &p01[o + nl0]);
        __builtin_nontemporal_store(__expf(e1 - rm) * ri, &p01[o + nl1]);
        __builtin_nontemporal_store(__expf(e0 - cm0) * ci0, &p10[o + nl0]);
        __builtin_nontemporal_store(__expf(e1 - cm1) * ci1, &p10[o + nl1]);
    }
}

// ---------------------------------------------------------------------------
// Kernel 4: exact fp32 logvar projections (D -> 2), one wave per row.
// ---------------------------------------------------------------------------
__global__ __launch_bounds__(256) void logvar_kernel(
    const float* __restrict__ desc0, const float* __restrict__ desc1,
    const float* __restrict__ Wl, const float* __restrict__ bl,
    float* __restrict__ lv01, float* __restrict__ lv10)
{
    const int tid = threadIdx.x;
    const int wv  = tid >> 5;
    const int ln  = tid & 31;
    const int row = blockIdx.x * 8 + wv;

    const float* src;
    float* dst;
    if (row < BB * MM) { src = desc0 + (size_t)row * DD; dst = lv01 + (size_t)row * 2; }
    else {
        int r2 = row - BB * MM;
        src = desc1 + (size_t)r2 * DD;
        dst = lv10 + (size_t)r2 * 2;
    }
    const float4* s4 = (const float4*)src;
    const float4* w0 = (const float4*)Wl;
    const float4* w1 = (const float4*)(Wl + DD);
    float a0 = 0.0f, a1 = 0.0f;
    #pragma unroll
    for (int i = 0; i < 2; ++i) {
        float4 x = s4[ln * 2 + i];
        float4 u = w0[ln * 2 + i];
        float4 t = w1[ln * 2 + i];
        a0 += x.x * u.x + x.y * u.y + x.z * u.z + x.w * u.w;
        a1 += x.x * t.x + x.y * t.y + x.z * t.z + x.w * t.w;
    }
    #pragma unroll
    for (int m = 16; m >= 1; m >>= 1) {
        a0 += __shfl_xor(a0, m, 32);
        a1 += __shfl_xor(a1, m, 32);
    }
    if (ln == 0) {
        dst[0] = a0 + bl[0];
        dst[1] = a1 + bl[1];
    }
}

// ---------------------------------------------------------------------------
extern "C" void kernel_launch(void* const* d_in, const int* in_sizes, int n_in,
                              void* d_out, int out_size, void* d_ws, size_t ws_size,
                              hipStream_t stream) {
    (void)in_sizes; (void)n_in; (void)out_size; (void)ws_size;

    const float* desc0 = (const float*)d_in[0];
    const float* desc1 = (const float*)d_in[1];
    const float* Wf    = (const float*)d_in[2];
    const float* bfv   = (const float*)d_in[3];
    const float* Wl    = (const float*)d_in[4];
    const float* bl    = (const float*)d_in[5];

    float* out  = (float*)d_out;
    float* p01  = out;
    float* p10  = out + (size_t)BB * MM * NN;
    float* lv01 = p10 + (size_t)BB * MM * NN;
    float* lv10 = lv01 + (size_t)BB * MM * 2;

    // Workspace: bf16 mdesc0/mdesc1 (16 MB, L2-resident) + softmax statistics.
    unsigned short* md0 = (unsigned short*)d_ws;
    unsigned short* md1 = md0 + (size_t)BB * MM * DD;
    float* rmx = (float*)(md1 + (size_t)BB * NN * DD);
    float* rin = rmx + BB * MM;
    float* cmx = rin + BB * MM;
    float* cin = cmx + BB * NN;

    proj_kernel<<<dim3(DD / 64, (BB * (MM + NN)) / 64), 256, 0, stream>>>(
        desc0, desc1, Wf, bfv, md0, md1);
    logvar_kernel<<<dim3((BB * (MM + NN)) / 8), 256, 0, stream>>>(
        desc0, desc1, Wl, bl, lv01, lv10);
    // Row-softmax stats of sim, then of sim^T (== column stats of sim).
    stats_kernel<<<dim3(BB * MM / 64), 256, 0, stream>>>(md0, md1, rmx, rin);
    stats_kernel<<<dim3(BB * NN / 64), 256, 0, stream>>>(md1, md0, cmx, cin);
    pwrite_kernel<<<dim3(NN / 64, MM / 64, BB), 256, 0, stream>>>(
        md0, md1, rmx, rin, cmx, cin, p01, p10);
}
